// HausdorffDistance_42082089566512
// MI455X (gfx1250) — matrix operand order repaired and checked by reading, not compiled, not run on hardware
//
#include <hip/hip_runtime.h>

// Hausdorff distance via fused tiled GEMM (bf16 split-precision WMMA) + min/max epilogue.
// MI455X / gfx1250: wave32, v_wmma_f32_16x16x32_bf16, 320KB LDS per WGP.
// v2: 32x32 wave tile (2x2 fragments) -> 12 WMMA per 16 ds_load_b128 per K-step
//     (1.5x better LDS-bytes/FLOP than v1's 16x32 wave tile).

typedef __attribute__((ext_vector_type(16))) __bf16 v16bf;
typedef __attribute__((ext_vector_type(8)))  float  v8f;

namespace {
constexpr int kB = 4;
constexpr int kN = 8192;
constexpr int kM = 8192;
constexpr int kD = 128;

constexpr int TN  = 128;  // output tile rows (X points): 4 row-wave groups x 32
constexpr int TM  = 64;   // output tile cols (Y points): 2 col-wave groups x 32
constexpr int KC  = 64;   // K chunk staged in LDS
constexpr int KST = 72;   // LDS row stride in bf16 elems (144B = 36 banks -> conflict-free b128 frags)
}

// ---- float <-> bf16 split helpers (manual RNE; hi + lo recovers ~fp32 accuracy) ----
__device__ __forceinline__ unsigned short f2bf(float f) {
  unsigned u = __float_as_uint(f);
  u += 0x7FFFu + ((u >> 16) & 1u);          // round-to-nearest-even
  return (unsigned short)(u >> 16);
}
__device__ __forceinline__ float bf2f(unsigned short h) {
  return __uint_as_float((unsigned)h << 16);
}
__device__ __forceinline__ void split_store4(unsigned short* hi, unsigned short* lo, float4 v) {
  unsigned short h0 = f2bf(v.x), h1 = f2bf(v.y), h2 = f2bf(v.z), h3 = f2bf(v.w);
  unsigned short l0 = f2bf(v.x - bf2f(h0)), l1 = f2bf(v.y - bf2f(h1));
  unsigned short l2 = f2bf(v.z - bf2f(h2)), l3 = f2bf(v.w - bf2f(h3));
  *(unsigned*)(hi)     = (unsigned)h0 | ((unsigned)h1 << 16);
  *(unsigned*)(hi + 2) = (unsigned)h2 | ((unsigned)h3 << 16);
  *(unsigned*)(lo)     = (unsigned)l0 | ((unsigned)l1 << 16);
  *(unsigned*)(lo + 2) = (unsigned)l2 | ((unsigned)l3 << 16);
}

// ---- monotonic unsigned encoding of float: enc(a) < enc(b) <=> a < b ----
__device__ __forceinline__ unsigned encf(float f) {
  unsigned u = __float_as_uint(f);
  return (u & 0x80000000u) ? ~u : (u | 0x80000000u);
}
__device__ __forceinline__ float decf(unsigned u) {
  return __uint_as_float((u & 0x80000000u) ? (u & 0x7FFFFFFFu) : ~u);
}

// ---- WMMA fragment loads from LDS (ISA 7.12.2 layouts) ----
// A 16x32 bf16: lane = (half<<4)|m ; VGPR0-3 = K[half*8 .. +7], VGPR4-7 = +16
__device__ __forceinline__ v16bf frag_a(const unsigned short* rowp, int k0) {
  union { uint4 q[2]; v16bf v; } f;
  f.q[0] = *(const uint4*)(rowp + k0);
  f.q[1] = *(const uint4*)(rowp + k0 + 16);
  return f.v;
}
// B 32x16 bf16: lane = (half<<4)|n ; VGPR0-3 = K[half*8 .. +7], VGPR4-7 = +16
__device__ __forceinline__ v16bf frag_b(const unsigned short* rowp, int k0) {
  union { uint4 q[2]; v16bf v; } f;
  f.q[0] = *(const uint4*)(rowp + k0);
  f.q[1] = *(const uint4*)(rowp + k0 + 16);
  return f.v;
}

// ================= kernel 1: exact fp32 squared norms (one wave per point) =================
__global__ __launch_bounds__(256) void HD_norm(const float* __restrict__ pts,
                                               float* __restrict__ out, int npts) {
  const int gw   = (int)((blockIdx.x * blockDim.x + threadIdx.x) >> 5);
  const int lane = threadIdx.x & 31;
  if (gw >= npts) return;
  float4 v = *((const float4*)(pts + (size_t)gw * kD) + lane);  // 32 lanes x 4 = 128 dims
  float s = v.x * v.x + v.y * v.y + v.z * v.z + v.w * v.w;
  s += __shfl_xor(s, 1, 32);
  s += __shfl_xor(s, 2, 32);
  s += __shfl_xor(s, 4, 32);
  s += __shfl_xor(s, 8, 32);
  s += __shfl_xor(s, 16, 32);
  if (lane == 0) out[gw] = s;
}

// ================= kernel 2: init encoded min arrays to enc(+inf) =================
__global__ __launch_bounds__(256) void HD_init(unsigned* __restrict__ p, int n) {
  int i = blockIdx.x * blockDim.x + threadIdx.x;
  if (i < n) p[i] = 0xFF800000u;  // encf(+inf)
}

// ================= kernel 3: fused tile GEMM + distance + row/col min =================
__global__ __launch_bounds__(256) void HD_tile(
    const float* __restrict__ X, const float* __restrict__ Y,
    const float* __restrict__ X2, const float* __restrict__ Y2,
    unsigned* __restrict__ rowU, unsigned* __restrict__ colU) {
  __shared__ unsigned short sXhi[TN * KST], sXlo[TN * KST];
  __shared__ unsigned short sYhi[TM * KST], sYlo[TM * KST];
  __shared__ unsigned sRow[TN], sCol[TM];

  const int t    = threadIdx.x;
  const int b    = blockIdx.z;
  const int row0 = blockIdx.y * TN;
  const int col0 = blockIdx.x * TM;

  if (t < TN) sRow[t] = 0xFFFFFFFFu;
  if (t < TM) sCol[t] = 0xFFFFFFFFu;

  const int wave = t >> 5, lane = t & 31;
  const int wrow = wave & 3;      // 4 row-wave groups of 32 rows (2 fragments)
  const int wcol = wave >> 2;     // 2 col-wave groups of 32 cols (2 fragments)
  const int l15 = lane & 15, half = lane >> 4;

  v8f acc[2][2];                  // [row frag][col frag]
  const v8f vzero = {0.f, 0.f, 0.f, 0.f, 0.f, 0.f, 0.f, 0.f};
#pragma unroll
  for (int rf = 0; rf < 2; ++rf)
#pragma unroll
    for (int c = 0; c < 2; ++c) acc[rf][c] = vzero;

  // staging maps: X: 2 threads/row x 32 floats ; Y: 4 threads/row x 16 floats
  const int xlr = t >> 1;         // 0..127
  const int xcb = (t & 1) << 5;   // 0 or 32
  const int ylr = t >> 2;         // 0..63
  const int ycb = (t & 3) << 4;   // 0,16,32,48

  for (int kc = 0; kc < kD; kc += KC) {
    if (kc) __syncthreads();  // previous chunk fully consumed before overwrite
    const float* xs = X + ((size_t)b * kN + row0 + xlr) * kD + kc + xcb;
    const float* ys = Y + ((size_t)b * kM + col0 + ylr) * kD + kc + ycb;
#pragma unroll
    for (int i = 0; i < 32; i += 4)
      split_store4(&sXhi[xlr * KST + xcb + i], &sXlo[xlr * KST + xcb + i], *(const float4*)(xs + i));
#pragma unroll
    for (int i = 0; i < 16; i += 4)
      split_store4(&sYhi[ylr * KST + ycb + i], &sYlo[ylr * KST + ycb + i], *(const float4*)(ys + i));
    __syncthreads();

#pragma unroll
    for (int ks = 0; ks < KC / 32; ++ks) {
      const int ka  = ks * 32 + half * 8;
      const int kb2 = ks * 32 + half * 8;
      v16bf ahi[2], alo[2], bhi[2], blo[2];
#pragma unroll
      for (int rf = 0; rf < 2; ++rf) {
        const int arow = wrow * 32 + rf * 16 + l15;
        ahi[rf] = frag_a(&sXhi[arow * KST], ka);
        alo[rf] = frag_a(&sXlo[arow * KST], ka);
      }
#pragma unroll
      for (int c = 0; c < 2; ++c) {
        const int brow = wcol * 32 + c * 16 + l15;
        bhi[c] = frag_b(&sYhi[brow * KST], kb2);
        blo[c] = frag_b(&sYlo[brow * KST], kb2);
      }
#pragma unroll
      for (int rf = 0; rf < 2; ++rf)
#pragma unroll
        for (int c = 0; c < 2; ++c) {
          v8f a = acc[rf][c];
          a = __builtin_amdgcn_wmma_f32_16x16x32_bf16(false, ahi[rf], false, bhi[c], (short)0, a, false, false);
          a = __builtin_amdgcn_wmma_f32_16x16x32_bf16(false, ahi[rf], false, blo[c], (short)0, a, false, false);
          a = __builtin_amdgcn_wmma_f32_16x16x32_bf16(false, alo[rf], false, bhi[c], (short)0, a, false, false);
          acc[rf][c] = a;
        }
    }
  }

  // ---- epilogue: d = |x|^2 + |y|^2 - 2*C ; row/col mins ----
  float x2v[2][8];
#pragma unroll
  for (int rf = 0; rf < 2; ++rf)
#pragma unroll
    for (int r = 0; r < 8; ++r)
      x2v[rf][r] = X2[(size_t)b * kN + row0 + wrow * 32 + rf * 16 + half * 8 + r];

  float rowm[2][8];
#pragma unroll
  for (int rf = 0; rf < 2; ++rf)
#pragma unroll
    for (int r = 0; r < 8; ++r) rowm[rf][r] = __builtin_inff();

#pragma unroll
  for (int c = 0; c < 2; ++c) {
    const int colg = col0 + wcol * 32 + c * 16 + l15;
    const float y2 = Y2[(size_t)b * kM + colg];
    float cm = __builtin_inff();
#pragma unroll
    for (int rf = 0; rf < 2; ++rf)
#pragma unroll
      for (int r = 0; r < 8; ++r) {
        float d = x2v[rf][r] + y2 - 2.0f * acc[rf][c][r];
        rowm[rf][r] = fminf(rowm[rf][r], d);
        cm = fminf(cm, d);
      }
    cm = fminf(cm, __shfl_xor(cm, 16, 32));  // combine both row-halves (same column)
    if (lane < 16) atomicMin(&sCol[wcol * 32 + c * 16 + l15], encf(cm));
  }

#pragma unroll
  for (int rf = 0; rf < 2; ++rf)
#pragma unroll
    for (int r = 0; r < 8; ++r) {
      float v = rowm[rf][r];
      v = fminf(v, __shfl_xor(v, 1, 32));
      v = fminf(v, __shfl_xor(v, 2, 32));
      v = fminf(v, __shfl_xor(v, 4, 32));
      v = fminf(v, __shfl_xor(v, 8, 32));    // min over 16 columns within the half-group
      if (l15 == 0) atomicMin(&sRow[wrow * 32 + rf * 16 + half * 8 + r], encf(v));
    }

  __syncthreads();
  if (t < TN)           atomicMin(&rowU[(size_t)b * kN + row0 + t], sRow[t]);
  else if (t < TN + TM) atomicMin(&colU[(size_t)b * kM + col0 + (t - TN)], sCol[t - TN]);
}

// ================= kernel 4: per-batch max of encoded mins =================
__global__ __launch_bounds__(256) void HD_final(const unsigned* __restrict__ rowU,
                                                const unsigned* __restrict__ colU,
                                                float* __restrict__ out) {
  __shared__ unsigned sr[256], sc[256];
  const int b = blockIdx.x, t = threadIdx.x;
  unsigned mr = 0u, mc = 0u;
  for (int i = t; i < kN; i += 256) { unsigned v = rowU[(size_t)b * kN + i]; mr = v > mr ? v : mr; }
  for (int i = t; i < kM; i += 256) { unsigned v = colU[(size_t)b * kM + i]; mc = v > mc ? v : mc; }
  sr[t] = mr; sc[t] = mc;
  __syncthreads();
  for (int s = 128; s > 0; s >>= 1) {
    if (t < s) {
      sr[t] = sr[t] > sr[t + s] ? sr[t] : sr[t + s];
      sc[t] = sc[t] > sc[t + s] ? sc[t] : sc[t + s];
    }
    __syncthreads();
  }
  if (t == 0) out[b] = fmaxf(decf(sr[0]), decf(sc[0]));  // encoding is monotonic: max commutes
}

extern "C" void kernel_launch(void* const* d_in, const int* in_sizes, int n_in,
                              void* d_out, int out_size, void* d_ws, size_t ws_size,
                              hipStream_t stream) {
  (void)in_sizes; (void)n_in; (void)out_size; (void)ws_size;
  const float* X = (const float*)d_in[0];
  const float* Y = (const float*)d_in[1];

  float*    ws   = (float*)d_ws;
  float*    X2   = ws;                                   // B*N floats
  float*    Y2   = ws + (size_t)kB * kN;                 // B*M floats
  unsigned* rowU = (unsigned*)(ws + 2 * (size_t)kB * kN);// B*N u32 (encoded mins)
  unsigned* colU = rowU + (size_t)kB * kN;               // B*M u32

  HD_norm<<<dim3((kB * kN) / 8), 256, 0, stream>>>(X, X2, kB * kN);
  HD_norm<<<dim3((kB * kM) / 8), 256, 0, stream>>>(Y, Y2, kB * kM);
  HD_init<<<dim3((2 * kB * kN) / 256), 256, 0, stream>>>(rowU, 2 * kB * kN);
  HD_tile<<<dim3(kM / TM, kN / TN, kB), 256, 0, stream>>>(X, Y, X2, Y2, rowU, colU);
  HD_final<<<dim3(kB), 256, 0, stream>>>(rowU, colU, (float*)d_out);
}
